// RecCore_74758200754205
// MI455X (gfx1250) — matrix operand order repaired and checked by reading, not compiled, not run on hardware
//
#include <hip/hip_runtime.h>
#include <hip/hip_bf16.h>

// Problem constants (from reference)
#define M_    256
#define T_    28
#define RNN_  1024
#define FEAT_ 2048
#define ATT_  512
#define FK_   28
#define KCAT_ 3072          // RNN_ + FEAT_ (fused gate GEMM K)
#define N4F_  8192          // 4*FEAT_

typedef __attribute__((ext_vector_type(16))) __bf16 v16bf;
typedef __attribute__((ext_vector_type(8)))  float  v8f;
typedef __attribute__((ext_vector_type(4)))  int    v4i;

// gfx1250 async global->LDS path (ASYNCcnt-tracked), with safe fallback
#if defined(__gfx1250__) && __has_builtin(__builtin_amdgcn_global_load_async_to_lds_b128) && \
    __has_builtin(__builtin_amdgcn_s_wait_asynccnt)
#define USE_ASYNC_LDS 1
#else
#define USE_ASYNC_LDS 0
#endif

__device__ __forceinline__ float sigm_(float x) { return 1.0f / (1.0f + __expf(-x)); }

// fp32 -> bf16 bits with round-to-nearest-even
__device__ __forceinline__ unsigned short f2bf_(float f) {
  unsigned int u = __builtin_bit_cast(unsigned int, f);
  u += 0x7FFFu + ((u >> 16) & 1u);
  return (unsigned short)(u >> 16);
}

// ---------------------------------------------------------------------------
// Pack an (Mrows x K) f32 row-major matrix into WMMA A-fragment tiles (16x32).
// ISA 16-bit A layout: lanes 0-15 hold K {0..7,16..23}, lanes 16-31 hold
// K {8..15,24..31}; 16 bf16 per lane per tile. Tile block = 512 bf16.
// Tile address: ((mt*ktTotal + ktOff + kt) * 512).
// ---------------------------------------------------------------------------
__global__ void pack_a_kernel(const float* __restrict__ src,
                              unsigned short* __restrict__ dst,
                              int K, int ktTotal, int ktOff, long long total) {
  long long gid = (long long)blockIdx.x * blockDim.x + threadIdx.x;
  if (gid >= total) return;
  int  k   = (int)(gid % K);
  long long row = gid / K;
  int  mt  = (int)(row >> 4), rin = (int)(row & 15);
  int  kt  = k >> 5, kin = k & 31;
  int  lane = rin + (((kin >> 3) & 1) ? 16 : 0);
  int  e    = (kin & 7) + ((kin & 16) ? 8 : 0);
  long long blk = ((long long)mt * ktTotal + ktOff + kt) << 9;
  dst[blk + lane * 16 + e] = f2bf_(src[gid]);
}

// ---------------------------------------------------------------------------
// Pack a (K x N) f32 row-major matrix into WMMA B-fragment tiles (32x16).
// ISA 16-bit B layout: lanes 0-15 hold K {0..7,16..23} for column n=lane, lanes
// 16-31 hold K {8..15,24..31}. Tile address: ((nt*ktTotal + ktOff + kt) * 512).
// ---------------------------------------------------------------------------
__global__ void pack_b_kernel(const float* __restrict__ src,
                              unsigned short* __restrict__ dst,
                              int N, int ktTotal, int ktOff, long long total) {
  long long gid = (long long)blockIdx.x * blockDim.x + threadIdx.x;
  if (gid >= total) return;
  int  n = (int)(gid % N);
  long long k = gid / N;
  int  kt = (int)(k >> 5), kin = (int)(k & 31);
  int  nt = n >> 4, nin = n & 15;
  int  lane = nin + (((kin >> 3) & 1) ? 16 : 0);
  int  e    = (kin & 7) + ((kin & 16) ? 8 : 0);
  long long blk = ((long long)nt * ktTotal + ktOff + kt) << 9;
  dst[blk + lane * 16 + e] = f2bf_(src[gid]);
}

// ---------------------------------------------------------------------------
// Fragment-layout bf16 GEMM, f32 accumulate: C[m,n] = sum_k A*B + bias[n].
// Block = 8 waves = 256 rows x 64 cols. Each wave owns a 32x64 tile
// (2 m-tiles x 4 n-tiles = 8 accumulators). The 4 B tiles of each K-step
// (4 KB, already in fragment layout) are staged once per block into LDS via
// GLOBAL_LOAD_ASYNC_TO_LDS_B128 and shared by all 8 waves (ds_load_b128),
// cutting B/L2 traffic 8x vs per-wave global loads. A streams from global.
// ---------------------------------------------------------------------------
__global__ void __launch_bounds__(256)
wmma_gemm_kernel(const unsigned short* __restrict__ Af,
                 const unsigned short* __restrict__ Bf,
                 const float* __restrict__ bias, float* __restrict__ C,
                 int N, int aKT, int aOff, int bKT, int bOff, int kCount) {
  __shared__ __align__(16) unsigned short btile[2048];  // 4 KB: 4 B tiles, fragment layout

  const int NB   = N >> 6;
  const int mblk = blockIdx.x / NB;          // 256-row block index
  const int nb   = blockIdx.x - mblk * NB;   // 64-col block index
  const int wave = threadIdx.x >> 5;
  const int lane = threadIdx.x & 31;
  const int mt0  = (mblk * 8 + wave) * 2;    // this wave's two 16-row tiles

  const v16bf* A0 = (const v16bf*)Af + (((long long)(mt0 + 0) * aKT + aOff) << 5);
  const v16bf* A1 = (const v16bf*)Af + (((long long)(mt0 + 1) * aKT + aOff) << 5);

  // Cooperative B staging: thread covers 16 B of tile tt (64 threads / 1KB tile)
  const int tt  = threadIdx.x >> 6;
  const int sub = (threadIdx.x & 63) << 4;
  const char* bsrc = (const char*)Bf + (((long long)(nb * 4 + tt) * bKT + bOff) << 10) + sub;
#if USE_ASYNC_LDS
  __attribute__((address_space(3))) char* ldst =
      (__attribute__((address_space(3))) char*)btile + tt * 1024 + sub;
#endif

  v8f acc[8];
#pragma unroll
  for (int i = 0; i < 8; ++i) acc[i] = (v8f){0.f, 0.f, 0.f, 0.f, 0.f, 0.f, 0.f, 0.f};

  const v16bf* Bl = (const v16bf*)btile;

  for (int k = 0; k < kCount; ++k) {
    __syncthreads();                         // previous iteration's LDS reads done
#if USE_ASYNC_LDS
    __builtin_amdgcn_global_load_async_to_lds_b128(
        (__attribute__((address_space(1))) v4i*)(bsrc + ((long long)k << 10)),
        (__attribute__((address_space(3))) v4i*)ldst, 0, 0);
    __builtin_amdgcn_s_wait_asynccnt(0);
#else
    *(uint4*)((char*)btile + tt * 1024 + sub) = *(const uint4*)(bsrc + ((long long)k << 10));
#endif
    __syncthreads();                         // staged B visible to all waves

    v16bf a0 = A0[(k << 5) + lane];
    v16bf a1 = A1[(k << 5) + lane];
    if (k + 1 < kCount) {                    // global_prefetch_b8 of next A tiles
      __builtin_prefetch((const void*)&A0[((k + 1) << 5) + lane], 0, 1);
      __builtin_prefetch((const void*)&A1[((k + 1) << 5) + lane], 0, 1);
    }
#pragma unroll
    for (int t = 0; t < 4; ++t) {
      v16bf bw = Bl[(t << 5) + lane];        // ds_load_b128 pair from staged tile
      acc[2 * t + 0] = __builtin_amdgcn_wmma_f32_16x16x32_bf16(
          false, a0, false, bw, (short)0, acc[2 * t + 0], false, false);
      acc[2 * t + 1] = __builtin_amdgcn_wmma_f32_16x16x32_bf16(
          false, a1, false, bw, (short)0, acc[2 * t + 1], false, false);
    }
  }

  // C/D layout: VGPR j, lanes 0-15 -> M=j, lanes 16-31 -> M=j+8; N = lane%16.
  const int ncol = (nb << 6) + (lane & 15);
#pragma unroll
  for (int mi = 0; mi < 2; ++mi) {
    const int rbase = ((mt0 + mi) << 4) + ((lane >> 4) << 3);
#pragma unroll
    for (int t = 0; t < 4; ++t) {
      float bv = bias[ncol + t * 16];
#pragma unroll
      for (int j = 0; j < 8; ++j) {
        C[(long long)(rbase + j) * N + ncol + t * 16] = acc[2 * t + mi][j] + bv;
      }
    }
  }
}

// ---------------------------------------------------------------------------
// scores[m,t] = sum_a tanh(hW[m,a] + va[m,t,a]) * Wa2w[a] + ba2w
// ---------------------------------------------------------------------------
__global__ void scores_kernel(const float* __restrict__ hW,
                              const float* __restrict__ va,
                              const float* __restrict__ Wa2w,
                              const float* __restrict__ ba2w,
                              float* __restrict__ scores) {
  int mt = blockIdx.x;          // m*T + t
  int m  = mt / T_;
  __shared__ float red[256];
  float p = 0.f;
  for (int a = threadIdx.x; a < ATT_; a += 256) {
    float e = tanhf(hW[m * ATT_ + a] + va[(long long)mt * ATT_ + a]);
    p += e * Wa2w[a];
  }
  red[threadIdx.x] = p;
  __syncthreads();
  for (int s = 128; s > 0; s >>= 1) {
    if (threadIdx.x < (unsigned)s) red[threadIdx.x] += red[threadIdx.x + s];
    __syncthreads();
  }
  if (threadIdx.x == 0) scores[mt] = red[0] + ba2w[0];
}

// ---------------------------------------------------------------------------
// softmax over T + weighted sum: dh_att[m,r] = sum_t alpha[m,t]*dh[m,t,r]
// ---------------------------------------------------------------------------
__global__ void attend_kernel(const float* __restrict__ scores,
                              const float* __restrict__ dh,
                              float* __restrict__ dh_att) {
  int m = blockIdx.x;
  float s[T_];
  float mx = -1e30f;
#pragma unroll
  for (int t = 0; t < T_; ++t) { s[t] = scores[m * T_ + t]; mx = fmaxf(mx, s[t]); }
  float sum = 0.f;
#pragma unroll
  for (int t = 0; t < T_; ++t) { s[t] = __expf(s[t] - mx); sum += s[t]; }
  float inv = 1.0f / sum;
#pragma unroll
  for (int t = 0; t < T_; ++t) s[t] *= inv;
  for (int r = threadIdx.x; r < RNN_; r += blockDim.x) {
    float acc = 0.f;
#pragma unroll
    for (int t = 0; t < T_; ++t)
      acc += s[t] * dh[((long long)m * T_ + t) * RNN_ + r];
    dh_att[m * RNN_ + r] = acc;
  }
}

// ---------------------------------------------------------------------------
// LSTM gates + state update + write h into output slab
// ---------------------------------------------------------------------------
__global__ void gates_kernel(const float* __restrict__ sums,
                             float* __restrict__ h, float* __restrict__ c,
                             float* __restrict__ out, int step) {
  long long gid = (long long)blockIdx.x * blockDim.x + threadIdx.x;
  if (gid >= (long long)M_ * FEAT_) return;
  int m = (int)(gid / FEAT_);
  int d = (int)(gid % FEAT_);
  const float* row = sums + (long long)m * N4F_;
  float ig = sigm_(row[d]);
  float fg = sigm_(row[FEAT_ + d]);
  float og = sigm_(row[2 * FEAT_ + d]);
  float g  = tanhf(row[3 * FEAT_ + d]);
  float c2 = fg * c[gid] + ig * g;
  float h2 = og * tanhf(c2);
  c[gid] = c2;
  h[gid] = h2;
  out[((long long)m * FK_ + step) * FEAT_ + d] = h2;
}

__global__ void bias_sum_kernel(const float* a, const float* b, float* o, int n) {
  int i = blockIdx.x * blockDim.x + threadIdx.x;
  if (i < n) o[i] = a[i] + b[i];
}

// ---------------------------------------------------------------------------
extern "C" void kernel_launch(void* const* d_in, const int* in_sizes, int n_in,
                              void* d_out, int out_size, void* d_ws, size_t ws_size,
                              hipStream_t stream) {
  (void)in_sizes; (void)n_in; (void)out_size; (void)ws_size;
  const float* dh   = (const float*)d_in[0];
  // d_in[1] = cap_mask (unused by the reference math)
  const float* h0   = (const float*)d_in[2];
  const float* c0   = (const float*)d_in[3];
  const float* Wi2h = (const float*)d_in[4];
  const float* bi2h = (const float*)d_in[5];
  const float* Wh2h = (const float*)d_in[6];
  const float* bh2h = (const float*)d_in[7];
  const float* Wh2a = (const float*)d_in[8];
  const float* bh2a = (const float*)d_in[9];
  const float* Wv2a = (const float*)d_in[10];
  const float* bv2a = (const float*)d_in[11];
  const float* Wa2w = (const float*)d_in[12];
  const float* ba2w = (const float*)d_in[13];
  float* out = (float*)d_out;

  // ---- workspace carve-up (256B aligned slabs) ----
  char* wsp = (char*)d_ws;
  auto carve = [&](size_t bytes) -> char* {
    char* p = wsp;
    wsp += (bytes + 255) & ~(size_t)255;
    return p;
  };
  unsigned short* wcatF  = (unsigned short*)carve((size_t)KCAT_ * N4F_ * 2);  // 48 MB
  unsigned short* wh2aF  = (unsigned short*)carve((size_t)FEAT_ * ATT_ * 2);  //  2 MB
  unsigned short* wv2aF  = (unsigned short*)carve((size_t)RNN_  * ATT_ * 2);  //  1 MB
  unsigned short* dhF    = (unsigned short*)carve((size_t)M_ * T_ * RNN_ * 2);// 14 MB
  unsigned short* acatF  = (unsigned short*)carve((size_t)M_ * KCAT_ * 2);    //1.5 MB
  float* va     = (float*)carve((size_t)M_ * T_ * ATT_ * 4);                  // 14 MB
  float* hW     = (float*)carve((size_t)M_ * ATT_ * 4);
  float* scores = (float*)carve((size_t)M_ * T_ * 4);
  float* dh_att = (float*)carve((size_t)M_ * RNN_ * 4);
  float* sums   = (float*)carve((size_t)M_ * N4F_ * 4);                       //  8 MB
  float* hbuf   = (float*)carve((size_t)M_ * FEAT_ * 4);
  float* cbuf   = (float*)carve((size_t)M_ * FEAT_ * 4);
  float* bsum   = (float*)carve((size_t)N4F_ * 4);

  const int TPB = 256;

  // ---- one-time packs (deterministic, re-done every call) ----
  {
    long long t1 = (long long)RNN_ * N4F_;           // Wi2h -> kt [0,32)
    pack_b_kernel<<<(unsigned)(t1 / TPB), TPB, 0, stream>>>(Wi2h, wcatF, N4F_, KCAT_ / 32, 0, t1);
    long long t2 = (long long)FEAT_ * N4F_;          // Wh2h -> kt [32,96)
    pack_b_kernel<<<(unsigned)(t2 / TPB), TPB, 0, stream>>>(Wh2h, wcatF, N4F_, KCAT_ / 32, 32, t2);
    long long t3 = (long long)FEAT_ * ATT_;
    pack_b_kernel<<<(unsigned)(t3 / TPB), TPB, 0, stream>>>(Wh2a, wh2aF, ATT_, FEAT_ / 32, 0, t3);
    long long t4 = (long long)RNN_ * ATT_;
    pack_b_kernel<<<(unsigned)(t4 / TPB), TPB, 0, stream>>>(Wv2a, wv2aF, ATT_, RNN_ / 32, 0, t4);
    long long t5 = (long long)M_ * T_ * RNN_;        // dh as A (7168 x 1024)
    pack_a_kernel<<<(unsigned)(t5 / TPB), TPB, 0, stream>>>(dh, dhF, RNN_, RNN_ / 32, 0, t5);
    bias_sum_kernel<<<N4F_ / TPB, TPB, 0, stream>>>(bi2h, bh2h, bsum, N4F_);
  }
  (void)hipMemcpyAsync(hbuf, h0, (size_t)M_ * FEAT_ * 4, hipMemcpyDeviceToDevice, stream);
  (void)hipMemcpyAsync(cbuf, c0, (size_t)M_ * FEAT_ * 4, hipMemcpyDeviceToDevice, stream);

  // ---- va = dh @ Wv2a + bv2a  (7168 x 512, K=1024) ----
  wmma_gemm_kernel<<<(M_ * T_ / 256) * (ATT_ / 64), TPB, 0, stream>>>(
      dhF, wv2aF, bv2a, va, ATT_, RNN_ / 32, 0, RNN_ / 32, 0, RNN_ / 32);

  // ---- 28 recurrent steps ----
  for (int step = 0; step < FK_; ++step) {
    // pack h into acat kt [32,96)
    long long th = (long long)M_ * FEAT_;
    pack_a_kernel<<<(unsigned)(th / TPB), TPB, 0, stream>>>(hbuf, acatF, FEAT_, KCAT_ / 32, 32, th);

    // hW = h @ Wh2a + bh2a   (256 x 512, K=2048)
    wmma_gemm_kernel<<<(M_ / 256) * (ATT_ / 64), TPB, 0, stream>>>(
        acatF, wh2aF, bh2a, hW, ATT_, KCAT_ / 32, 32, FEAT_ / 32, 0, FEAT_ / 32);

    // attention scores, softmax + weighted sum
    scores_kernel<<<M_ * T_, TPB, 0, stream>>>(hW, va, Wa2w, ba2w, scores);
    attend_kernel<<<M_, TPB, 0, stream>>>(scores, dh, dh_att);

    // pack dh_att into acat kt [0,32)
    long long ta = (long long)M_ * RNN_;
    pack_a_kernel<<<(unsigned)(ta / TPB), TPB, 0, stream>>>(dh_att, acatF, RNN_, KCAT_ / 32, 0, ta);

    // sums = [dh_att | h] @ [Wi2h ; Wh2h] + (bi2h + bh2h)   (256 x 8192, K=3072)
    wmma_gemm_kernel<<<(M_ / 256) * (N4F_ / 64), TPB, 0, stream>>>(
        acatF, wcatF, bsum, sums, N4F_, KCAT_ / 32, 0, KCAT_ / 32, 0, KCAT_ / 32);

    // gate nonlinearity + state update + output write
    gates_kernel<<<(unsigned)((long long)M_ * FEAT_ / TPB), TPB, 0, stream>>>(
        sums, hbuf, cbuf, out, step);
  }
}
